// xLSTM_3126736192251
// MI455X (gfx1250) — matrix operand order repaired and checked by reading, not compiled, not run on hardware
//
#include <hip/hip_runtime.h>
#include <math.h>

typedef _Float16 f16;
typedef f16   v16h __attribute__((ext_vector_type(16)));
typedef float v8f  __attribute__((ext_vector_type(8)));

#define Bb   8
#define Ss   512
#define Ein  512
#define Ee   1024
#define Dd   2048
#define NHh  4
#define DHm  512
#define DHs  256
#define UPf  1344
#define TOK  (Bb*Ss)
#define BSE  ((size_t)TOK*Ee)

// Async global->LDS DMA (gfx1250). Guarded so absence degrades to manual copy.
#if defined(__AMDGCN__) && __has_builtin(__builtin_amdgcn_global_load_async_to_lds_b128)
#define USE_ASYNC_LDS 1
#else
#define USE_ASYNC_LDS 0
#endif

#if USE_ASYNC_LDS
typedef int v4i_vs __attribute__((vector_size(16)));   // matches builtin pointee type
__device__ __forceinline__ void async_ld128(const float* g, float* l){
  __builtin_amdgcn_global_load_async_to_lds_b128(
      (__attribute__((address_space(1))) v4i_vs*)g,
      (__attribute__((address_space(3))) v4i_vs*)l,
      0, 0);
}
__device__ __forceinline__ void wait_async0(){
#if __has_builtin(__builtin_amdgcn_s_wait_asynccnt)
  __builtin_amdgcn_s_wait_asynccnt(0);
#else
  asm volatile("s_wait_asynccnt 0" ::: "memory");
#endif
}
#endif

__device__ __forceinline__ float siluf(float x){ return x/(1.f+__expf(-x)); }
__device__ __forceinline__ float sigmf(float x){ return 1.f/(1.f+__expf(-x)); }
__device__ __forceinline__ float logsigf(float x){ return fminf(x,0.f) - log1pf(__expf(-fabsf(x))); }
__device__ __forceinline__ float gelu_t(float x){
  float x3 = x*x*x;
  return 0.5f*x*(1.f+tanhf(0.7978845608028654f*(x+0.044715f*x3)));
}

// ---------------------------------------------------------------------------
// Generic WMMA GEMM:  C[m, z*zCoff + n] = act( sum_k A[m*lda + z*zAoff + k] *
//                       W[z*zWoff + n*K + k]  + bias[z*bzs + n] )
// Block tile 64x64, BK=32, 8 waves (4 m x 2 n), each wave 16x32 of C.
// Tiles staged to LDS in fp32 (async-DMA capable); cvt to f16 at fragment
// build. All M,N,K used are multiples of 64/64/32 -> no edge masking.
// ---------------------------------------------------------------------------
__global__ __launch_bounds__(256)
void gemm_wmma(const float* __restrict__ A, const float* __restrict__ W,
               const float* __restrict__ bias, float* __restrict__ C,
               int K, int lda, int ldc, int zAoff, int zWoff, int zCoff,
               int bzs, int act)
{
  __shared__ __align__(16) float As[64][32];
  __shared__ __align__(16) float Bs[64][32];
  const int tid  = threadIdx.x;
  const int z    = blockIdx.z;
  const int n0   = blockIdx.x*64, m0 = blockIdx.y*64;
  const int wave = tid>>5, lane = tid&31;
  const int wm   = wave>>1, wn = wave&1;
  const int lm   = lane&15, hi = lane>>4;

  const float* Az = A + (size_t)z*zAoff;
  const float* Wz = W + (size_t)z*zWoff;
  float*       Cz = C + (size_t)z*zCoff;

  v8f acc0 = {0.f,0.f,0.f,0.f,0.f,0.f,0.f,0.f};
  v8f acc1 = {0.f,0.f,0.f,0.f,0.f,0.f,0.f,0.f};

  for (int k0=0; k0<K; k0+=32){
    // ---- stage 64x32 fp32 tiles of A and W into LDS ----------------------
#if USE_ASYNC_LDS
    #pragma unroll
    for (int j=0;j<2;j++){
      int f = tid + j*256;            // float4 id 0..511
      int r = f>>3, c4 = (f&7)*4;
      async_ld128(Az + (size_t)(m0+r)*lda + k0 + c4, &As[r][c4]);
      async_ld128(Wz + (size_t)(n0+r)*K  + k0 + c4, &Bs[r][c4]);
    }
#else
    #pragma unroll
    for (int j=0;j<2;j++){
      int f = tid + j*256;
      int r = f>>3, c4 = (f&7)*4;
      *(float4*)&As[r][c4] = *(const float4*)(Az + (size_t)(m0+r)*lda + k0 + c4);
      *(float4*)&Bs[r][c4] = *(const float4*)(Wz + (size_t)(n0+r)*K  + k0 + c4);
    }
#endif
    // warm L2 for the next K-tile while this one lands
    if (k0 + 32 < K){
      __builtin_prefetch(Az + (size_t)(m0 + (tid>>2))*lda + k0 + 32 + (tid&3)*8, 0, 3);
      __builtin_prefetch(Wz + (size_t)(n0 + (tid>>2))*K  + k0 + 32 + (tid&3)*8, 0, 3);
    }
#if USE_ASYNC_LDS
    wait_async0();
#endif
    __syncthreads();

    // ---- fragments (contiguous k = hi*16 + s for A and B), cvt fp32->f16 -------------------
    v16h a, b0, b1;
    const float* ar  = &As[wm*16 + lm][0];
    const int    kb  = hi*16;
    #pragma unroll
    for (int j=0;j<8;j++){ a[j]=(f16)ar[kb+j]; a[8+j]=(f16)ar[kb+8+j]; }
    const float* br0 = &Bs[wn*32 + lm][0];
    const float* br1 = &Bs[wn*32 + 16 + lm][0];
    const int    kb2 = hi*16;
    #pragma unroll
    for (int j=0;j<16;j++){ b0[j]=(f16)br0[kb2+j]; b1[j]=(f16)br1[kb2+j]; }
    acc0 = __builtin_amdgcn_wmma_f32_16x16x32_f16(false,a,false,b0,(short)0,acc0,false,false);
    acc1 = __builtin_amdgcn_wmma_f32_16x16x32_f16(false,a,false,b1,(short)0,acc1,false,false);
    __syncthreads();
  }

  #pragma unroll
  for (int v=0; v<8; v++){
    int m  = m0 + wm*16 + v + hi*8;
    int n1 = n0 + wn*32 + lm;
    int n2 = n1 + 16;
    float x0 = acc0[v], x1 = acc1[v];
    if (bias){ x0 += bias[z*bzs + n1]; x1 += bias[z*bzs + n2]; }
    if (act==1){ x0=fmaxf(x0,0.f); x1=fmaxf(x1,0.f); }
    else if (act==2){ x0=siluf(x0); x1=siluf(x1); }
    Cz[(size_t)m*ldc + n1] = x0;
    Cz[(size_t)m*ldc + n2] = x1;
  }
}

// ---------------------------------------------------------------------------
// LayerNorm over last dim n, weight w. One block per row.
// ---------------------------------------------------------------------------
__global__ __launch_bounds__(256)
void ln_k(const float* __restrict__ x, const float* __restrict__ w,
          float* __restrict__ y, int n){
  int row = blockIdx.x, tid = threadIdx.x;
  const float* xr = x + (size_t)row*n;
  __shared__ float red[256];
  float s=0; for (int i=tid;i<n;i+=256) s+=xr[i];
  red[tid]=s; __syncthreads();
  for (int st=128;st>0;st>>=1){ if(tid<st) red[tid]+=red[tid+st]; __syncthreads(); }
  float mean = red[0]/n; __syncthreads();
  float v=0; for (int i=tid;i<n;i+=256){ float d=xr[i]-mean; v+=d*d; }
  red[tid]=v; __syncthreads();
  for (int st=128;st>0;st>>=1){ if(tid<st) red[tid]+=red[tid+st]; __syncthreads(); }
  float rstd = rsqrtf(red[0]/n + 1e-5f);
  float* yr = y + (size_t)row*n;
  for (int i=tid;i<n;i+=256) yr[i] = (xr[i]-mean)*rstd*w[i];
}

// ---------------------------------------------------------------------------
// Multi-head LayerNorm: per (row, head) over dh; w indexed [h*dh+d].
// ---------------------------------------------------------------------------
__global__ __launch_bounds__(256)
void mh_ln_k(const float* __restrict__ x, const float* __restrict__ w,
             float* __restrict__ y, int nh, int dh){
  int row = blockIdx.x, tid = threadIdx.x;
  int dim = nh*dh, tph = 256/nh;
  int hh = tid/tph, lt = tid%tph;
  const float* xr = x + (size_t)row*dim + hh*dh;
  __shared__ float red[256];
  float s=0; for (int d=lt; d<dh; d+=tph) s += xr[d];
  red[tid]=s; __syncthreads();
  for (int st=tph>>1; st>0; st>>=1){ if (lt<st) red[tid]+=red[tid+st]; __syncthreads(); }
  float mean = red[hh*tph]/dh; __syncthreads();
  float v=0; for (int d=lt; d<dh; d+=tph){ float t=xr[d]-mean; v+=t*t; }
  red[tid]=v; __syncthreads();
  for (int st=tph>>1; st>0; st>>=1){ if (lt<st) red[tid]+=red[tid+st]; __syncthreads(); }
  float rstd = rsqrtf(red[hh*tph]/dh + 1e-5f);
  float* yr = y + (size_t)row*dim + hh*dh;
  for (int d=lt; d<dh; d+=tph) yr[d] = (xr[d]-mean)*rstd*w[hh*dh+d];
}

// ---------------------------------------------------------------------------
// Depthwise causal conv1d (K=4) + SiLU. x has row stride xstride, C channels.
// ---------------------------------------------------------------------------
__global__ void conv_silu_k(const float* __restrict__ x, int xstride,
                            const float* __restrict__ w, const float* __restrict__ cb,
                            float* __restrict__ y, int C, long total){
  long i = (long)blockIdx.x*256 + threadIdx.x;
  if (i>=total) return;
  int  c   = (int)(i % C);
  long t   = i / C;               // token index b*Ss+s
  int  s   = (int)(t % Ss);
  long tok0= t - s;               // b*Ss
  float acc = cb[c];
  #pragma unroll
  for (int j=0;j<4;j++){
    int sp = s - 3 + j;
    if (sp>=0) acc += w[c*4+j]*x[(tok0+sp)*(long)xstride + c];
  }
  y[t*(long)C + c] = siluf(acc);
}

// ---------------------------------------------------------------------------
// Block-diagonal 4x4 qkv projections: q,k from xc (stride D); v from xm (stride 2D).
// ---------------------------------------------------------------------------
__global__ void qkv_k(const float* __restrict__ xc, const float* __restrict__ xm,
                      const float* __restrict__ qw, const float* __restrict__ kw,
                      const float* __restrict__ vw,
                      float* __restrict__ q, float* __restrict__ k, float* __restrict__ v){
  long i = (long)blockIdx.x*256 + threadIdx.x;
  long total = (long)TOK*(Dd/4);
  if (i>=total) return;
  int  nb  = (int)(i % (Dd/4));
  long tok = i/(Dd/4);
  float xcv[4], xmv[4];
  #pragma unroll
  for (int j=0;j<4;j++){
    xcv[j] = xc[tok*(long)Dd   + nb*4 + j];
    xmv[j] = xm[tok*(2L*Dd)    + nb*4 + j];
  }
  #pragma unroll
  for (int o=0;o<4;o++){
    float qa=0,ka=0,va=0;
    #pragma unroll
    for (int j=0;j<4;j++){
      qa += xcv[j]*qw[nb*16 + o*4 + j];
      ka += xcv[j]*kw[nb*16 + o*4 + j];
      va += xmv[j]*vw[nb*16 + o*4 + j];
    }
    q[tok*(long)Dd + nb*4 + o]=qa;
    k[tok*(long)Dd + nb*4 + o]=ka;
    v[tok*(long)Dd + nb*4 + o]=va;
  }
}

// ---------------------------------------------------------------------------
// Input/forget gate pre-activations: dot(qkv_concat, w[h]) + b[h]. Block/token.
// ---------------------------------------------------------------------------
__global__ __launch_bounds__(256)
void gates_k(const float* __restrict__ q, const float* __restrict__ k,
             const float* __restrict__ v,
             const float* __restrict__ igw, const float* __restrict__ igb,
             const float* __restrict__ fgw, const float* __restrict__ fgb,
             float* __restrict__ IG, float* __restrict__ FG){
  int tok = blockIdx.x, tid = threadIdx.x;
  int b = tok / Ss, s = tok % Ss;
  float ai[4]={0,0,0,0}, af[4]={0,0,0,0};
  for (int f=tid; f<Dd; f+=256){
    float qv=q[(size_t)tok*Dd+f], kv=k[(size_t)tok*Dd+f], vv=v[(size_t)tok*Dd+f];
    #pragma unroll
    for (int h=0;h<4;h++){
      const float* wI = igw + h*3*Dd;
      const float* wF = fgw + h*3*Dd;
      ai[h] += qv*wI[f] + kv*wI[Dd+f] + vv*wI[2*Dd+f];
      af[h] += qv*wF[f] + kv*wF[Dd+f] + vv*wF[2*Dd+f];
    }
  }
  __shared__ float red[256];
  for (int h=0;h<4;h++){
    red[tid]=ai[h]; __syncthreads();
    for (int st=128;st>0;st>>=1){ if(tid<st) red[tid]+=red[tid+st]; __syncthreads(); }
    if (tid==0) IG[((size_t)b*NHh+h)*Ss + s] = red[0] + igb[h];
    __syncthreads();
    red[tid]=af[h]; __syncthreads();
    for (int st=128;st>0;st>>=1){ if(tid<st) red[tid]+=red[tid+st]; __syncthreads(); }
    if (tid==0) FG[((size_t)b*NHh+h)*Ss + s] = red[0] + fgb[h];
    __syncthreads();
  }
}

// ---------------------------------------------------------------------------
// lf = cumsum(logsigmoid(fg)); maxD[s] = lf[s] + prefixmax(ig[t]-lf[t]).
// ---------------------------------------------------------------------------
__global__ void lfscan_k(const float* __restrict__ IG, const float* __restrict__ FG,
                         float* __restrict__ LF, float* __restrict__ MX){
  int t = blockIdx.x*blockDim.x + threadIdx.x;
  if (t >= Bb*NHh) return;
  size_t base = (size_t)t*Ss;
  float lf=0.f, pmx=-1e30f;
  for (int s=0;s<Ss;s++){
    lf += logsigf(FG[base+s]);
    float a = IG[base+s] - lf;
    pmx = fmaxf(pmx, a);
    LF[base+s]=lf;
    MX[base+s]=lf+pmx;
  }
}

// ---------------------------------------------------------------------------
// mLSTM parallel attention, WMMA, single pass (maxD precomputed).
// grid (S/16, NH, B), block 256 (8 waves). q,k,v,out: (B,S,NH,DH) = (B,S,D).
// ---------------------------------------------------------------------------
__global__ __launch_bounds__(256)
void mlstm_attn_k(const float* __restrict__ q, const float* __restrict__ k,
                  const float* __restrict__ v, const float* __restrict__ LF,
                  const float* __restrict__ IG, const float* __restrict__ MX,
                  float* __restrict__ out){
  __shared__ f16   qs[16][512];
  __shared__ f16   ps[16][128];
  __shared__ float den[16];
  __shared__ float denp[8][16];
  const int s0 = blockIdx.x*16, h = blockIdx.y, b = blockIdx.z;
  const int tid = threadIdx.x, wave = tid>>5, lane = tid&31;
  const int lm = lane&15, hi = lane>>4;
  const float scale = rsqrtf((float)DHm);

  // stage scaled q tile (16 x 512) into LDS as f16
  const float* qbase = q + ((size_t)b*Ss + s0)*Dd + h*DHm;
  #pragma unroll
  for (int j=0;j<8;j++){
    int idx4 = tid + j*256;           // 2048 float4s
    int r = idx4>>7, c4 = idx4&127;
    const float4 qv = *(const float4*)(qbase + (size_t)r*Dd + c4*4);
    qs[r][c4*4+0] = (f16)(qv.x*scale);
    qs[r][c4*4+1] = (f16)(qv.y*scale);
    qs[r][c4*4+2] = (f16)(qv.z*scale);
    qs[r][c4*4+3] = (f16)(qv.w*scale);
  }
  if (tid<16) den[tid]=0.f;

  const size_t vb = ((size_t)b*NHh + h)*Ss;
  float lf_row[8], md_row[8];
  #pragma unroll
  for (int vv=0; vv<8; vv++){
    int row = s0 + vv + hi*8;
    lf_row[vv] = LF[vb+row];
    md_row[vv] = MX[vb+row];
  }
  v8f acc[4];
  #pragma unroll
  for (int nt=0;nt<4;nt++) acc[nt] = (v8f){0.f,0.f,0.f,0.f,0.f,0.f,0.f,0.f};
  __syncthreads();

  const int tend = s0 + 16;
  const int dw   = wave*64;
  for (int tb0=0; tb0<tend; tb0+=128){
    // ---- phase 1: this wave computes one 16x16 S-tile (keys tt..tt+15) ----
    int tt = tb0 + wave*16;
    if (tt < tend){
      v8f sacc = {0.f,0.f,0.f,0.f,0.f,0.f,0.f,0.f};
      int tg = tt + lm;
      const float* kbp = k + ((size_t)b*Ss + tg)*Dd + h*DHm + hi*16;
      for (int kk=0; kk<DHm; kk+=32){
        v16h a, bfr;
        const f16* ar = &qs[lm][kk];
        const int  kb = hi*16;
        #pragma unroll
        for (int j=0;j<8;j++){ a[j]=ar[kb+j]; a[8+j]=ar[kb+8+j]; }
        const float4 k0v = *(const float4*)(kbp + kk);
        const float4 k1v = *(const float4*)(kbp + kk + 4);
        const float4 k2v = *(const float4*)(kbp + kk + 8);
        const float4 k3v = *(const float4*)(kbp + kk + 12);
        bfr[0]=(f16)k0v.x; bfr[1]=(f16)k0v.y; bfr[2]=(f16)k0v.z; bfr[3]=(f16)k0v.w;
        bfr[4]=(f16)k1v.x; bfr[5]=(f16)k1v.y; bfr[6]=(f16)k1v.z; bfr[7]=(f16)k1v.w;
        bfr[8]=(f16)k2v.x; bfr[9]=(f16)k2v.y; bfr[10]=(f16)k2v.z; bfr[11]=(f16)k2v.w;
        bfr[12]=(f16)k3v.x; bfr[13]=(f16)k3v.y; bfr[14]=(f16)k3v.z; bfr[15]=(f16)k3v.w;
        sacc = __builtin_amdgcn_wmma_f32_16x16x32_f16(false,a,false,bfr,(short)0,sacc,false,false);
      }
      float lf_t = LF[vb+tg], ig_t = IG[vb+tg];
      #pragma unroll
      for (int vv=0; vv<8; vv++){
        int sg = s0 + vv + hi*8;
        float p = (tg <= sg) ? sacc[vv]*__expf(lf_row[vv]-lf_t+ig_t-md_row[vv]) : 0.f;
        float rs = p;
        #pragma unroll
        for (int off=1; off<16; off<<=1) rs += __shfl_xor(rs, off, 32);
        if (lm==0) denp[wave][vv+hi*8] = rs;
        ps[vv+hi*8][wave*16 + lm] = (f16)p;
      }
    } else {
      #pragma unroll
      for (int vv=0; vv<8; vv++){
        ps[vv+hi*8][wave*16+lm] = (f16)0.f;
        if (lm==0) denp[wave][vv+hi*8] = 0.f;
      }
    }
    __syncthreads();
    // deterministic normalizer accumulation
    if (tid<16){
      float s=0.f;
      #pragma unroll
      for (int w8=0; w8<8; w8++) s += denp[w8][tid];
      den[tid] += s;
    }
    // ---- phase 2: every wave accumulates its 64-column DH slice -----------
    for (int st=0; st<4; st++){
      if (tb0 + st*32 >= tend) break;
      v16h a;
      const f16* pr = &ps[lm][st*32];
      const int  kb = hi*16;
      #pragma unroll
      for (int j=0;j<8;j++){ a[j]=pr[kb+j]; a[8+j]=pr[kb+8+j]; }
      #pragma unroll
      for (int nt=0; nt<4; nt++){
        v16h bfr;
        int d = dw + nt*16 + lm;
        const float* vbp = v + ((size_t)b*Ss + tb0 + st*32 + hi*16)*Dd + h*DHm + d;
        #pragma unroll
        for (int j=0;j<16;j++) bfr[j] = (f16)vbp[(size_t)j*Dd];
        acc[nt] = __builtin_amdgcn_wmma_f32_16x16x32_f16(false,a,false,bfr,(short)0,acc[nt],false,false);
      }
    }
    __syncthreads();
  }
  // epilogue: h = num / (max(|den|, exp(-maxD)) + 1e-6)
  #pragma unroll
  for (int vv=0; vv<8; vv++){
    float nrm = fmaxf(fabsf(den[vv+hi*8]), __expf(-md_row[vv])) + 1e-6f;
    int m = s0 + vv + hi*8;
    #pragma unroll
    for (int nt=0; nt<4; nt++){
      out[((size_t)b*Ss + m)*Dd + h*DHm + dw + nt*16 + lm] = acc[nt][vv]/nrm;
    }
  }
}

// ---------------------------------------------------------------------------
// hs = (hn + skip*xc) * silu(z),  z = xz[:, D:2D]
// ---------------------------------------------------------------------------
__global__ void hs_k(const float* __restrict__ hn, const float* __restrict__ xc,
                     const float* __restrict__ skip, const float* __restrict__ xz,
                     float* __restrict__ out, long total){
  long i = (long)blockIdx.x*256 + threadIdx.x;
  if (i>=total) return;
  int  c   = (int)(i % Dd);
  long tok = i / Dd;
  float z = xz[tok*(2L*Dd) + Dd + c];
  out[i] = (hn[i] + skip[c]*xc[i]) * siluf(z);
}

__global__ void add_k(float* __restrict__ y, const float* __restrict__ t, long total){
  long i = (long)blockIdx.x*256 + threadIdx.x;
  if (i<total) y[i] += t[i];
}

__global__ void gelumul_k(const float* __restrict__ up, float* __restrict__ out, long total){
  long i = (long)blockIdx.x*256 + threadIdx.x;
  if (i>=total) return;
  int  j   = (int)(i % UPf);
  long tok = i / UPf;
  float g = up[tok*(2L*UPf) + j];
  float u = up[tok*(2L*UPf) + UPf + j];
  out[i] = gelu_t(g)*u;
}

// ---------------------------------------------------------------------------
// sLSTM sequential scan. One block per (b,head), thread = hidden index k.
// pre: (4,B,S,E) head-interleaved; Rg: (4,NH,DH,DH); ys: (B,S,E).
// ---------------------------------------------------------------------------
__global__ __launch_bounds__(256)
void slstm_scan_k(const float* __restrict__ pre, const float* __restrict__ Rg,
                  float* __restrict__ ys){
  int b = blockIdx.x, n = blockIdx.y, kidx = threadIdx.x;
  __shared__ __align__(16) float hsh[DHs];
  const float* R0 = Rg + (((size_t)(0*NHh + n))*DHs + kidx)*DHs;
  const float* R1 = Rg + (((size_t)(1*NHh + n))*DHs + kidx)*DHs;
  const float* R2 = Rg + (((size_t)(2*NHh + n))*DHs + kidx)*DHs;
  const float* R3 = Rg + (((size_t)(3*NHh + n))*DHs + kidx)*DHs;
  float c=0.f, nn=0.f, mm=0.f;
  hsh[kidx]=0.f;
  __syncthreads();
  for (int s=0;s<Ss;s++){
    float r0=0,r1=0,r2=0,r3=0;
    for (int d=0; d<DHs; d+=4){
      float4 hv = *(const float4*)&hsh[d];
      float4 w0 = *(const float4*)&R0[d];
      float4 w1 = *(const float4*)&R1[d];
      float4 w2 = *(const float4*)&R2[d];
      float4 w3 = *(const float4*)&R3[d];
      r0 += hv.x*w0.x + hv.y*w0.y + hv.z*w0.z + hv.w*w0.w;
      r1 += hv.x*w1.x + hv.y*w1.y + hv.z*w1.z + hv.w*w1.w;
      r2 += hv.x*w2.x + hv.y*w2.y + hv.z*w2.z + hv.w*w2.w;
      r3 += hv.x*w3.x + hv.y*w3.y + hv.z*w3.z + hv.w*w3.w;
    }
    size_t base = ((size_t)b*Ss + s)*Ee + n*DHs + kidx;
    float ib = pre[base]         + r0;
    float fb = pre[BSE + base]   + r1;
    float zb = pre[2*BSE + base] + r2;
    float ob = pre[3*BSE + base] + r3;
    float lgf  = logsigf(fb);
    float mnew = fmaxf(ib, lgf + mm);
    float iv = __expf(ib - mnew);
    float fv = __expf(lgf + mm - mnew);
    c  = fv*c  + iv*tanhf(zb);
    nn = fv*nn + iv;
    float hk = sigmf(ob) * c / (nn + 1e-6f);
    mm = mnew;
    __syncthreads();
    hsh[kidx] = hk;
    ys[base]  = hk;
    __syncthreads();
  }
}

// ---------------------------------------------------------------------------
extern "C" void kernel_launch(void* const* d_in, const int* in_sizes, int n_in,
                              void* d_out, int out_size, void* d_ws, size_t ws_size,
                              hipStream_t stream){
  (void)in_sizes; (void)n_in; (void)out_size; (void)ws_size;
  const float* x         = (const float*)d_in[0];
  const float* ffn_in_w  = (const float*)d_in[1];
  const float* ffn_in_b  = (const float*)d_in[2];
  const float* m_ln_w    = (const float*)d_in[3];
  const float* proj_up_w = (const float*)d_in[4];
  const float* m_conv_w  = (const float*)d_in[5];
  const float* m_conv_b  = (const float*)d_in[6];
  const float* q_w       = (const float*)d_in[7];
  const float* k_w       = (const float*)d_in[8];
  const float* v_w       = (const float*)d_in[9];
  const float* ig_w      = (const float*)d_in[10];
  const float* ig_b      = (const float*)d_in[11];
  const float* fg_w      = (const float*)d_in[12];
  const float* fg_b      = (const float*)d_in[13];
  const float* mhn_w_m   = (const float*)d_in[14];
  const float* skip_w    = (const float*)d_in[15];
  const float* proj_dn_w = (const float*)d_in[16];
  const float* ln1_w     = (const float*)d_in[17];
  const float* s_conv_w  = (const float*)d_in[18];
  const float* s_conv_b  = (const float*)d_in[19];
  const float* Wg        = (const float*)d_in[20];
  const float* Rg        = (const float*)d_in[21];
  const float* bg        = (const float*)d_in[22];
  const float* mhn_w_s   = (const float*)d_in[23];
  const float* ln2_w     = (const float*)d_in[24];
  const float* ffn_up_w  = (const float*)d_in[25];
  const float* ffn_dn_w  = (const float*)d_in[26];
  const float* post_ln_w = (const float*)d_in[27];
  float* out = (float*)d_out;
  float* ws  = (float*)d_ws;

  const size_t M4 = (size_t)TOK*Ee;      // 4M floats
  float* H    = ws;                      // residual (B,S,E)
  float* Abuf = ws + 1*M4;               // LN outputs
  float* T    = ws + 2*M4;               // temp (B,S,E)
  float* BIG  = ws + 3*M4;               // 4*M4: xz / pre / ffn_up
  float* XC   = ws + 7*M4;               // 2*M4: conv outs / gelu*u
  float* Q    = ws + 9*M4;               // 2*M4
  float* Kb   = ws + 11*M4;              // 2*M4
  float* V    = ws + 13*M4;              // 2*M4
  float* HM   = ws + 15*M4;              // 2*M4: attention out
  float* IG   = ws + 17*M4;
  float* FG   = IG + (size_t)Bb*NHh*Ss;
  float* LF   = FG + (size_t)Bb*NHh*Ss;
  float* MX   = LF + (size_t)Bb*NHh*Ss;

  dim3 blk(256);
  const long totE  = (long)TOK*Ee;
  const long totD  = (long)TOK*Dd;
  const long totNB = (long)TOK*(Dd/4);
  const long totU  = (long)TOK*UPf;

  // h = relu(x @ ffn_in^T + b)
  gemm_wmma<<<dim3(Ee/64, TOK/64, 1), blk, 0, stream>>>(x, ffn_in_w, ffn_in_b, H,
        Ein, Ein, Ee, 0,0,0, 0, 1);
  // ---- mLSTM block ----
  ln_k<<<TOK, blk, 0, stream>>>(H, m_ln_w, Abuf, Ee);
  gemm_wmma<<<dim3(2*Dd/64, TOK/64, 1), blk, 0, stream>>>(Abuf, proj_up_w, nullptr, BIG,
        Ee, Ee, 2*Dd, 0,0,0, 0, 0);
  conv_silu_k<<<(unsigned)((totD+255)/256), blk, 0, stream>>>(BIG, 2*Dd, m_conv_w, m_conv_b, XC, Dd, totD);
  qkv_k<<<(unsigned)((totNB+255)/256), blk, 0, stream>>>(XC, BIG, q_w, k_w, v_w, Q, Kb, V);
  gates_k<<<TOK, blk, 0, stream>>>(Q, Kb, V, ig_w, ig_b, fg_w, fg_b, IG, FG);
  lfscan_k<<<1, 32, 0, stream>>>(IG, FG, LF, MX);
  mlstm_attn_k<<<dim3(Ss/16, NHh, Bb), blk, 0, stream>>>(Q, Kb, V, LF, IG, MX, HM);
  mh_ln_k<<<TOK, blk, 0, stream>>>(HM, mhn_w_m, Q, NHh, DHm);          // hn -> Q
  hs_k<<<(unsigned)((totD+255)/256), blk, 0, stream>>>(Q, XC, skip_w, BIG, Kb, totD);
  gemm_wmma<<<dim3(Ee/64, TOK/64, 1), blk, 0, stream>>>(Kb, proj_dn_w, nullptr, T,
        Dd, Dd, Ee, 0,0,0, 0, 0);
  add_k<<<(unsigned)((totE+255)/256), blk, 0, stream>>>(H, T, totE);
  // ---- sLSTM block ----
  ln_k<<<TOK, blk, 0, stream>>>(H, ln1_w, Abuf, Ee);
  conv_silu_k<<<(unsigned)((totE+255)/256), blk, 0, stream>>>(Abuf, Ee, s_conv_w, s_conv_b, XC, Ee, totE);
  for (int g=0; g<4; g++){
    const float* Ain = (g<2) ? XC : Abuf;
    gemm_wmma<<<dim3(DHs/64, TOK/64, NHh), blk, 0, stream>>>(
        Ain, Wg + (size_t)g*NHh*DHs*DHs, bg + (size_t)g*NHh*DHs,
        BIG + (size_t)g*M4,
        DHs, Ee, Ee, DHs, DHs*DHs, DHs, DHs, 0);
  }
  slstm_scan_k<<<dim3(Bb, NHh), dim3(DHs), 0, stream>>>(BIG, Rg, V);
  mh_ln_k<<<TOK, blk, 0, stream>>>(V, mhn_w_s, T, NHh, DHs);
  add_k<<<(unsigned)((totE+255)/256), blk, 0, stream>>>(H, T, totE);
  // ---- gated FFN ----
  ln_k<<<TOK, blk, 0, stream>>>(H, ln2_w, Abuf, Ee);
  gemm_wmma<<<dim3(2*UPf/64, TOK/64, 1), blk, 0, stream>>>(Abuf, ffn_up_w, nullptr, BIG,
        Ee, Ee, 2*UPf, 0,0,0, 0, 0);
  gelumul_k<<<(unsigned)((totU+255)/256), blk, 0, stream>>>(BIG, XC, totU);
  gemm_wmma<<<dim3(Ee/64, TOK/64, 1), blk, 0, stream>>>(XC, ffn_dn_w, nullptr, T,
        UPf, UPf, Ee, 0,0,0, 0, 0);
  add_k<<<(unsigned)((totE+255)/256), blk, 0, stream>>>(H, T, totE);
  // ---- post norm -> out ----
  ln_k<<<TOK, blk, 0, stream>>>(H, post_ln_w, out, Ee);
}
